// ContextAttention_29953101923145
// MI455X (gfx1250) — matrix operand order repaired and checked by reading, not compiled, not run on hardware
//
#include <hip/hip_runtime.h>
#include <hip/hip_bf16.h>

typedef _Float16 v16h __attribute__((ext_vector_type(16)));
typedef float    v8f  __attribute__((ext_vector_type(8)));

#define CB 4
#define CC 64
#define CH 64
#define CW 64
#define HW 4096
#define KP 576          // CC * 9
#define ROWB (KP * 2)   // P row stride in bytes = 1152

// ---- fragment load helpers (byte offsets; base already includes hi-lane skew) ----
// A fragment (16x32 f16): 32 contiguous bytes
__device__ __forceinline__ v16h load_fragA(const char* base, int byteoff) {
  v16h r;
  *((uint4*)&r)     = *(const uint4*)(base + byteoff);
  *((uint4*)&r + 1) = *(const uint4*)(base + byteoff + 16);
  return r;
}
// B fragment (32x16 f16): 32 contiguous bytes
__device__ __forceinline__ v16h load_fragB(const char* base, int byteoff) {
  v16h r;
  *((uint4*)&r)     = *(const uint4*)(base + byteoff);
  *((uint4*)&r + 1) = *(const uint4*)(base + byteoff + 16);
  return r;
}
// 2x4 tile of WMMAs
__device__ __forceinline__ void mma8(v8f acc[2][4], const v16h a[2], const v16h b[4]) {
  #pragma unroll
  for (int mf = 0; mf < 2; ++mf)
    #pragma unroll
    for (int nf = 0; nf < 4; ++nf)
      acc[mf][nf] = __builtin_amdgcn_wmma_f32_16x16x32_f16(
          false, a[mf], false, b[nf], (short)0, acc[mf][nf], false, false);
}

// ---------------- kernel: per-(b,c) inverse spatial L2 norm ----------------
__global__ __launch_bounds__(256) void k_norm(const float* __restrict__ x,
                                              float* __restrict__ invn) {
  __shared__ float red[256];
  const int bc = blockIdx.x;                         // 0..255 = b*64+c
  const float* px = x + (size_t)bc * HW;
  float s = 0.f;
  for (int i = threadIdx.x; i < HW; i += 256) { float v = px[i]; s += v * v; }
  red[threadIdx.x] = s; __syncthreads();
  for (int st = 128; st > 0; st >>= 1) {
    if (threadIdx.x < st) red[threadIdx.x] += red[threadIdx.x + st];
    __syncthreads();
  }
  if (threadIdx.x == 0) invn[bc] = rsqrtf(red[0] + 1e-8f);
}

// ---------------- kernel: background mask ----------------------------------
__global__ __launch_bounds__(256) void k_bkg(const float* __restrict__ mask,
                                             float* __restrict__ bkg) {
  const int idx = blockIdx.x * 256 + threadIdx.x;    // 0..16383
  const int b = idx >> 12, p = idx & (HW - 1);
  const int y = p >> 6, xx = p & 63;
  float any = 0.f;
  #pragma unroll
  for (int dy = -1; dy <= 1; ++dy)
    #pragma unroll
    for (int dx = -1; dx <= 1; ++dx) {
      const int ny = y + dy, nx = xx + dx;
      if ((unsigned)ny < CH && (unsigned)nx < CW)
        if (mask[(size_t)b * HW + ny * CW + nx] < 0.5f) any = 1.f;
    }
  bkg[idx] = any;
}

// ---------------- kernel: pre-swizzled B operand for GEMM2 -----------------
// Bsw[b][kt][nf][lane][j] (f16), value = x[b][c][q]*bkg[b][q],
// q = kt*32 + 16*(lane/16) + j, c = nf*16 + lane%16  (B operand; contiguous k order, same as A)
__global__ __launch_bounds__(256) void k_bsw(const float* __restrict__ x,
                                             const float* __restrict__ bkg,
                                             _Float16* __restrict__ Bsw) {
  const int idx = blockIdx.x * 256 + threadIdx.x;    // 20 bits
  const int j    =  idx        & 15;
  const int lane = (idx >> 4)  & 31;
  const int nf   = (idx >> 9)  & 3;
  const int kt   = (idx >> 11) & 127;
  const int b    =  idx >> 18;
  const int q = kt * 32 + ((lane >> 4) << 4) + j;
  const int c = nf * 16 + (lane & 15);
  const float v = x[((size_t)b * CC + c) * HW + q] * bkg[(size_t)b * HW + q];
  Bsw[idx] = (_Float16)v;
}

// ---------------- kernel: build f16 patch matrix P[p][c*9+t] (one batch) ---
__global__ __launch_bounds__(256) void k_patch(const float* __restrict__ xb,
                                               const float* __restrict__ invn_b,
                                               _Float16* __restrict__ P) {
  const int idx = blockIdx.x * 256 + threadIdx.x;    // 0..262143
  const int p = idx & (HW - 1);
  const int c = idx >> 12;
  const float inv = invn_b[c];
  const int y = p >> 6, xx = p & 63;
  _Float16* row = P + (size_t)p * KP + c * 9;
  #pragma unroll
  for (int t = 0; t < 9; ++t) {
    const int ny = y + t / 3 - 1, nx = xx + t % 3 - 1;
    float v = 0.f;
    if ((unsigned)ny < CH && (unsigned)nx < CW) v = xb[c * HW + ny * CW + nx] * inv;
    row[t] = (_Float16)v;
  }
}

// ---------------- kernel: GEMM1  G = P * P^T  (4096x4096, K=576) -----------
// 128x128 block tile, 8 waves of 32x64 (2x4 WMMA frags), K fully unrolled with
// register double-buffering so each K-tile's loads issue before the previous
// tile's WMMAs.
__global__ __launch_bounds__(256) void k_gemm1(const _Float16* __restrict__ P,
                                               float* __restrict__ G) {
  const int lane = threadIdx.x & 31;
  const int wave = threadIdx.x >> 5;
  const int wm = wave & 3, wn = wave >> 2;
  const int hi = lane >> 4, lo = lane & 15;
  const int p0 = blockIdx.x * 128 + wm * 32;   // M base of this wave
  const int q0 = blockIdx.y * 128 + wn * 64;   // N base of this wave

  const char* baseA[2];
  const char* baseB[4];
  #pragma unroll
  for (int mf = 0; mf < 2; ++mf)
    baseA[mf] = (const char*)P + (size_t)(p0 + mf * 16 + lo) * ROWB + 32 * hi;
  #pragma unroll
  for (int nf = 0; nf < 4; ++nf)
    baseB[nf] = (const char*)P + (size_t)(q0 + nf * 16 + lo) * ROWB + 32 * hi;

  v8f  acc[2][4] = {};
  v16h a[2][2], bf[2][4];

  // preload K-tile 0 into buffer 0 (byte offset per tile = 64)
  #pragma unroll
  for (int mf = 0; mf < 2; ++mf) a[0][mf] = load_fragA(baseA[mf], 0);
  #pragma unroll
  for (int nf = 0; nf < 4; ++nf) bf[0][nf] = load_fragB(baseB[nf], 0);

  #pragma unroll
  for (int kt = 0; kt < 18; kt += 2) {
    // prefetch tile kt+1 into buffer 1
    #pragma unroll
    for (int mf = 0; mf < 2; ++mf) a[1][mf] = load_fragA(baseA[mf], (kt + 1) * 64);
    #pragma unroll
    for (int nf = 0; nf < 4; ++nf) bf[1][nf] = load_fragB(baseB[nf], (kt + 1) * 64);
    mma8(acc, a[0], bf[0]);
    if (kt + 2 < 18) {
      // prefetch tile kt+2 into buffer 0
      #pragma unroll
      for (int mf = 0; mf < 2; ++mf) a[0][mf] = load_fragA(baseA[mf], (kt + 2) * 64);
      #pragma unroll
      for (int nf = 0; nf < 4; ++nf) bf[0][nf] = load_fragB(baseB[nf], (kt + 2) * 64);
    }
    mma8(acc, a[1], bf[1]);
  }

  #pragma unroll
  for (int mf = 0; mf < 2; ++mf)
    #pragma unroll
    for (int nf = 0; nf < 4; ++nf) {
      const int col = q0 + nf * 16 + lo;
      #pragma unroll
      for (int v = 0; v < 8; ++v) {
        const int row = p0 + mf * 16 + hi * 8 + v;
        G[(size_t)row * HW + col] = acc[mf][nf][v];
      }
    }
}

// ---------------- kernel: row softmax of G -> f16 S ------------------------
__global__ __launch_bounds__(256) void k_softmax(const float* __restrict__ G,
                                                 _Float16* __restrict__ S) {
  __shared__ float red[256];
  const int t = threadIdx.x;
  const size_t row = (size_t)blockIdx.x * HW;
  const float4* r4 = (const float4*)(G + row);
  float4 f[4];
  float m = -3.402823466e38f;
  #pragma unroll
  for (int i = 0; i < 4; ++i) {
    f[i] = r4[i * 256 + t];
    m = fmaxf(m, fmaxf(fmaxf(f[i].x, f[i].y), fmaxf(f[i].z, f[i].w)));
  }
  red[t] = m; __syncthreads();
  for (int st = 128; st > 0; st >>= 1) {
    if (t < st) red[t] = fmaxf(red[t], red[t + st]);
    __syncthreads();
  }
  const float mx = red[0];
  __syncthreads();
  float sum = 0.f;
  #pragma unroll
  for (int i = 0; i < 4; ++i) {
    f[i].x = __expf(f[i].x - mx); f[i].y = __expf(f[i].y - mx);
    f[i].z = __expf(f[i].z - mx); f[i].w = __expf(f[i].w - mx);
    sum += f[i].x + f[i].y + f[i].z + f[i].w;
  }
  red[t] = sum; __syncthreads();
  for (int st = 128; st > 0; st >>= 1) {
    if (t < st) red[t] += red[t + st];
    __syncthreads();
  }
  const float inv = 1.0f / red[0];
  #pragma unroll
  for (int i = 0; i < 4; ++i) {
    union { _Float16 h[4]; uint2 u; } pk;
    pk.h[0] = (_Float16)(f[i].x * inv); pk.h[1] = (_Float16)(f[i].y * inv);
    pk.h[2] = (_Float16)(f[i].z * inv); pk.h[3] = (_Float16)(f[i].w * inv);
    *(uint2*)(S + row + (size_t)(i * 256 + t) * 4) = pk.u;
  }
}

// ---------------- kernel: GEMM2  Y[p][c] = S[p][q] * (x*bkg)[q][c] ---------
// 8 waves stacked in M (block tile 256x64), double-buffered fragments.
__global__ __launch_bounds__(256) void k_gemm2(const _Float16* __restrict__ S,
                                               const _Float16* __restrict__ Bsw,
                                               float* __restrict__ Y) {
  const int lane = threadIdx.x & 31;
  const int wave = threadIdx.x >> 5;
  const int hi = lane >> 4, lo = lane & 15;
  const int p0 = blockIdx.x * 256 + wave * 32;

  const char* baseA[2];
  #pragma unroll
  for (int mf = 0; mf < 2; ++mf)
    baseA[mf] = (const char*)S + (size_t)(p0 + mf * 16 + lo) * (HW * 2) + 32 * hi;
  const char* baseB[4];
  #pragma unroll
  for (int nf = 0; nf < 4; ++nf)
    baseB[nf] = (const char*)Bsw + (size_t)(nf * 32 + lane) * 32;
  // per K-tile strides (bytes): A advances 64, B advances 4096

  v8f  acc[2][4] = {};
  v16h a[2][2], bf[2][4];

  #pragma unroll
  for (int mf = 0; mf < 2; ++mf) a[0][mf] = load_fragA(baseA[mf], 0);
  #pragma unroll
  for (int nf = 0; nf < 4; ++nf) bf[0][nf] = load_fragB(baseB[nf], 0);

  for (int kt = 0; kt < 128; kt += 2) {
    #pragma unroll
    for (int mf = 0; mf < 2; ++mf) a[1][mf] = load_fragA(baseA[mf], (kt + 1) * 64);
    #pragma unroll
    for (int nf = 0; nf < 4; ++nf) bf[1][nf] = load_fragB(baseB[nf], (kt + 1) * 4096);
    mma8(acc, a[0], bf[0]);
    if (kt + 2 < 128) {
      #pragma unroll
      for (int mf = 0; mf < 2; ++mf) a[0][mf] = load_fragA(baseA[mf], (kt + 2) * 64);
      #pragma unroll
      for (int nf = 0; nf < 4; ++nf) bf[0][nf] = load_fragB(baseB[nf], (kt + 2) * 4096);
    }
    mma8(acc, a[1], bf[1]);
  }

  #pragma unroll
  for (int mf = 0; mf < 2; ++mf)
    #pragma unroll
    for (int nf = 0; nf < 4; ++nf) {
      const int col = nf * 16 + lo;
      #pragma unroll
      for (int v = 0; v < 8; ++v) {
        const int row = p0 + mf * 16 + hi * 8 + v;
        Y[(size_t)row * CC + col] = acc[mf][nf][v];
      }
    }
}

// ---------------- kernel: 3x3 reflect box smooth + mask, write output ------
__device__ __forceinline__ int refl64(int v) {
  return v < 0 ? -v : (v > 63 ? 126 - v : v);
}
__global__ __launch_bounds__(256) void k_smooth(const float* __restrict__ Y,
                                                const float* __restrict__ mask_b,
                                                float* __restrict__ out_b) {
  const int idx = blockIdx.x * 256 + threadIdx.x;    // 0..262143
  const int c = idx & 63;
  const int p = idx >> 6;
  const int y = p >> 6, xx = p & 63;
  float acc = 0.f;
  #pragma unroll
  for (int dy = -1; dy <= 1; ++dy) {
    const int ny = refl64(y + dy);
    #pragma unroll
    for (int dx = -1; dx <= 1; ++dx) {
      const int nx = refl64(xx + dx);
      acc += Y[(size_t)(ny * CW + nx) * CC + c];
    }
  }
  out_b[(size_t)c * HW + p] = mask_b[p] * acc * (1.f / 9.f);
}

// ---------------- host launch ----------------------------------------------
extern "C" void kernel_launch(void* const* d_in, const int* in_sizes, int n_in,
                              void* d_out, int out_size, void* d_ws, size_t ws_size,
                              hipStream_t stream) {
  (void)in_sizes; (void)n_in; (void)out_size; (void)ws_size;
  const float* x    = (const float*)d_in[0];   // [4,64,64,64]
  const float* mask = (const float*)d_in[1];   // [4,1,64,64]
  float* out = (float*)d_out;                  // [4,64,64,64]

  char* ws = (char*)d_ws;
  size_t off = 0;
  float*    invn = (float*)(ws + off);    off += 1024;                       // 256 f32
  float*    bkg  = (float*)(ws + off);    off += (size_t)CB * HW * 4;        // 64 KB
  _Float16* Bsw  = (_Float16*)(ws + off); off += (size_t)CB * HW * CC * 2;   // 2 MB
  _Float16* P    = (_Float16*)(ws + off); off += (size_t)HW * KP * 2;        // 4.5 MB (per-batch reuse)
  float*    G    = (float*)(ws + off);    off += (size_t)HW * HW * 4;        // 64 MB  (per-batch reuse; fits L2)
  _Float16* Sh   = (_Float16*)(ws + off); off += (size_t)HW * HW * 2;        // 32 MB  (per-batch reuse)
  float*    Y    = (float*)(ws + off);    off += (size_t)HW * CC * 4;        // 1 MB   (per-batch reuse)

  k_norm<<<CB * CC, 256, 0, stream>>>(x, invn);
  k_bkg <<<(CB * HW) / 256, 256, 0, stream>>>(mask, bkg);
  k_bsw <<<(CB * HW * CC) / 256, 256, 0, stream>>>(x, bkg, Bsw);

  for (int b = 0; b < CB; ++b) {
    k_patch<<<(HW * CC) / 256, 256, 0, stream>>>(x + (size_t)b * CC * HW,
                                                 invn + b * CC, P);
    dim3 g1(HW / 128, HW / 128);
    k_gemm1<<<g1, 256, 0, stream>>>(P, G);
    k_softmax<<<HW, 256, 0, stream>>>(G, Sh);
    k_gemm2<<<HW / 256, 256, 0, stream>>>(Sh, Bsw + (size_t)b * HW * CC, Y);
    k_smooth<<<(HW * CC) / 256, 256, 0, stream>>>(Y, mask + (size_t)b * HW,
                                                  out + (size_t)b * CC * HW);
  }
}
